// DualAlignedConv2d_23141283791480
// MI455X (gfx1250) — matrix operand order repaired and checked by reading, not compiled, not run on hardware
//
#include <hip/hip_runtime.h>
#include <hip/hip_bf16.h>
#include <cstdint>
#include <cstddef>

// Problem constants (match reference)
#define BB    4
#define H1    64
#define W1    64
#define POS   (H1*W1)     // 4096
#define FD    64
#define RADI  4
#define KIN   96          // conv1 K: 83 channels padded to 3*32
#define HID   64
#define ITERS 20
#define HH    256
#define WW    256

typedef __attribute__((ext_vector_type(16))) __bf16       v16bf;
typedef __attribute__((ext_vector_type(8)))  float        v8f;
typedef __attribute__((ext_vector_type(8)))  unsigned int v8u;

union ABf { v8u u; v16bf h; };

__device__ __forceinline__ uint16_t f2bf(float f) {
  uint32_t u = __float_as_uint(f);
  u += 0x7FFFu + ((u >> 16) & 1u);          // round-to-nearest-even
  return (uint16_t)(u >> 16);
}
__device__ __forceinline__ float bf2f(uint16_t h) {
  return __uint_as_float(((uint32_t)h) << 16);
}
__device__ __forceinline__ uint32_t pack2(float a, float b) {
  return (uint32_t)f2bf(a) | ((uint32_t)f2bf(b) << 16);
}
__device__ __forceinline__ void zeroF(ABf& a) {
  #pragma unroll
  for (int v = 0; v < 8; ++v) a.u[v] = 0u;
}

// A-operand fragment (16-bit A 16x32 layout): two 8-element chunks per lane,
// K = khalf*8 + [0,8) and 16 + khalf*8 + [0,8), khalf = lane>>4.
__device__ __forceinline__ void loadA(ABf& a, const uint16_t* row, int half) {
  uint4 c0 = *(const uint4*)(row + half * 8);
  uint4 c1 = *(const uint4*)(row + 16 + half * 8);
  a.u[0]=c0.x; a.u[1]=c0.y; a.u[2]=c0.z; a.u[3]=c0.w;
  a.u[4]=c1.x; a.u[5]=c1.y; a.u[6]=c1.z; a.u[7]=c1.w;
}
// B-operand fragment (16-bit B 32x16 layout): two 8-element chunks per lane,
// K = half*8 + [0,8) and 16 + half*8 + [0,8) of its column row.
__device__ __forceinline__ void loadB(ABf& b, const uint16_t* row, int half) {
  uint4 c0 = *(const uint4*)(row + half * 8);
  uint4 c1 = *(const uint4*)(row + 16 + half * 8);
  b.u[0]=c0.x; b.u[1]=c0.y; b.u[2]=c0.z; b.u[3]=c0.w;
  b.u[4]=c1.x; b.u[5]=c1.y; b.u[6]=c1.z; b.u[7]=c1.w;
}

// ---------------------------------------------------------------------------
// Encoder: 8x8 stride-4 conv, 2ch -> 64ch.  Both outputs row-major bf16
// [b][pos][64] (A-side array for f1; B columns read the same layout).
// blockIdx.y==0: sd0 -> fA.  blockIdx.y==1: sd1 -> fB.
// ---------------------------------------------------------------------------
__global__ void __launch_bounds__(256)
enc_kernel(const float* __restrict__ x, const float* __restrict__ ew,
           const float* __restrict__ eb,
           uint16_t* __restrict__ fA, uint16_t* __restrict__ fB) {
  __shared__ float w[FD * 128];             // [oc][ic*64 + ky*8 + kx], 32 KB
  for (int i = threadIdx.x; i < FD * 128; i += blockDim.x) w[i] = ew[i];
  __syncthreads();

  int gid = blockIdx.x * blockDim.x + threadIdx.x;   // BB*POS threads
  int b = gid >> 12, pos = gid & 4095;
  int y1 = pos >> 6, x1 = pos & 63;
  int mode  = blockIdx.y;
  int cbase = mode ? 5 : 7;                 // sd1 = x[:,5:7], sd0 = x[:,7:9]

  float acc[FD];
  #pragma unroll
  for (int oc = 0; oc < FD; ++oc) acc[oc] = eb[oc];

  for (int ic = 0; ic < 2; ++ic) {
    const float* xp = x + ((size_t)b * 9 + cbase + ic) * HH * WW;
    for (int ky = 0; ky < 8; ++ky) {
      int iy = y1 * 4 - 2 + ky;
      if (iy < 0 || iy >= HH) continue;
      for (int kx = 0; kx < 8; ++kx) {
        int ix = x1 * 4 - 2 + kx;
        if (ix < 0 || ix >= WW) continue;
        float v = xp[iy * WW + ix];
        const float* wr = w + ic * 64 + ky * 8 + kx;
        #pragma unroll
        for (int oc = 0; oc < FD; ++oc) acc[oc] = fmaf(v, wr[oc * 128], acc[oc]);
      }
    }
  }

  uint16_t* o = (mode == 0 ? fA : fB) + (size_t)gid * FD;
  uint4* o4 = (uint4*)o;
  #pragma unroll
  for (int q = 0; q < 8; ++q) {
    uint4 s;
    s.x = pack2(acc[8*q+0], acc[8*q+1]);
    s.y = pack2(acc[8*q+2], acc[8*q+3]);
    s.z = pack2(acc[8*q+4], acc[8*q+5]);
    s.w = pack2(acc[8*q+6], acc[8*q+7]);
    o4[q] = s;
  }
}

// ---------------------------------------------------------------------------
// Pack uw1 [64][83][3][3] f32 -> wf [tap][oc=64][k=96] bf16 (zero-padded K).
// ---------------------------------------------------------------------------
__global__ void wf_kernel(const float* __restrict__ uw1, uint16_t* __restrict__ wf) {
  int gid = blockIdx.x * blockDim.x + threadIdx.x;
  if (gid >= 9 * 64 * KIN) return;
  int t = gid / (64 * KIN);
  int r = gid - t * 64 * KIN;
  int n = r / KIN, k = r - n * KIN;
  int ky = t / 3, kx = t - ky * 3;
  float v = (k < 83) ? uw1[((n * 83 + k) * 3 + ky) * 3 + kx] : 0.f;
  wf[gid] = f2bf(v);
}

// ---------------------------------------------------------------------------
// Correlation GEMM: per batch, [4096,64] x [64,4096] -> bf16 corr (x 1/8).
// One wave computes a 32x64 strip: 4 A frags, 2 B frags per N-tile, 16 WMMAs.
// ---------------------------------------------------------------------------
__global__ void __launch_bounds__(256)
corr_kernel(const uint16_t* __restrict__ fA, const uint16_t* __restrict__ fB,
            uint16_t* __restrict__ cor) {
  int wid  = (blockIdx.x * 256 + (int)threadIdx.x) >> 5;
  int lane = threadIdx.x & 31;
  int half = lane >> 4, m = lane & 15;
  int b  = wid >> 13;             // 8192 waves per batch
  int r2 = wid & 8191;
  int mt = r2 >> 6;               // 32-row M tile (0..127)
  int ng = r2 & 63;               // 64-col N group

  ABf a[4];                       // [mhalf][kchunk]
  #pragma unroll
  for (int hr = 0; hr < 2; ++hr) {
    const uint16_t* ar = fA + ((size_t)b * POS + mt * 32 + hr * 16 + m) * FD;
    loadA(a[hr * 2 + 0], ar, half);
    loadA(a[hr * 2 + 1], ar + 32, half);
  }

  v8f acc[8] = {};                // [mhalf*4 + nt]
  #pragma unroll
  for (int nt = 0; nt < 4; ++nt) {
    const uint16_t* br = fB + ((size_t)b * POS + ng * 64 + nt * 16 + m) * FD;
    ABf b0, b1;
    loadB(b0, br, half);          // K 0..31
    loadB(b1, br + 32, half);     // K 32..63
    acc[nt] = __builtin_amdgcn_wmma_f32_16x16x32_bf16(
        false, a[0].h, false, b0.h, (short)0, acc[nt], false, false);
    acc[nt] = __builtin_amdgcn_wmma_f32_16x16x32_bf16(
        false, a[1].h, false, b1.h, (short)0, acc[nt], false, false);
    acc[4 + nt] = __builtin_amdgcn_wmma_f32_16x16x32_bf16(
        false, a[2].h, false, b0.h, (short)0, acc[4 + nt], false, false);
    acc[4 + nt] = __builtin_amdgcn_wmma_f32_16x16x32_bf16(
        false, a[3].h, false, b1.h, (short)0, acc[4 + nt], false, false);
  }

  #pragma unroll
  for (int hr = 0; hr < 2; ++hr) {
    #pragma unroll
    for (int nt = 0; nt < 4; ++nt) {
      uint16_t* orow = cor + ((size_t)b * POS + mt * 32 + hr * 16 + half * 8) * POS
                           + ng * 64 + nt * 16 + m;
      #pragma unroll
      for (int r = 0; r < 8; ++r)
        orow[(size_t)r * POS] = f2bf(acc[hr * 4 + nt][r] * 0.125f);  // /sqrt(64)
    }
  }
}

// ---------------------------------------------------------------------------
// Bilinear sample of the 81-point correlation window -> bf16 conv1 input
// rows [b][pos][96]; packed register staging, 12 b128 stores per pixel.
// ---------------------------------------------------------------------------
__device__ __forceinline__ float bsamp(const uint16_t* __restrict__ s,
                                       float xs, float ys) {
  float xf = floorf(xs), yf = floorf(ys);
  float wx1 = xs - xf, wy1 = ys - yf;
  int xi = (int)xf, yi = (int)yf;
  float v00 = 0.f, v10 = 0.f, v01 = 0.f, v11 = 0.f;
  if ((unsigned)yi < 64u) {
    if ((unsigned)xi       < 64u) v00 = bf2f(s[yi * 64 + xi]);
    if ((unsigned)(xi + 1) < 64u) v10 = bf2f(s[yi * 64 + xi + 1]);
  }
  if ((unsigned)(yi + 1) < 64u) {
    if ((unsigned)xi       < 64u) v01 = bf2f(s[(yi + 1) * 64 + xi]);
    if ((unsigned)(xi + 1) < 64u) v11 = bf2f(s[(yi + 1) * 64 + xi + 1]);
  }
  float wx0 = 1.f - wx1, wy0 = 1.f - wy1;
  return (v00 * wx0 + v10 * wx1) * wy0 + (v01 * wx0 + v11 * wx1) * wy1;
}

__global__ void __launch_bounds__(256)
sample_kernel(const uint16_t* __restrict__ cor, const float* __restrict__ flow,
              uint16_t* __restrict__ uin) {
  int gid = blockIdx.x * blockDim.x + threadIdx.x;   // BB*POS
  int b = gid >> 12, pos = gid & 4095;
  int y = pos >> 6, xq = pos & 63;
  float fx = flow[((size_t)b * 2 + 0) * POS + pos];
  float fy = flow[((size_t)b * 2 + 1) * POS + pos];
  float cx = xq + fx, cy = y + fy;
  const uint16_t* s = cor + (size_t)gid * POS;

  float t[KIN];
  #pragma unroll
  for (int dy = 0; dy < 9; ++dy) {
    #pragma unroll
    for (int dx = 0; dx < 9; ++dx)
      t[dy * 9 + dx] = bsamp(s, cx + (float)(dx - RADI), cy + (float)(dy - RADI));
  }
  t[81] = fx; t[82] = fy;
  #pragma unroll
  for (int k = 83; k < KIN; ++k) t[k] = 0.f;

  uint4* u4 = (uint4*)(uin + (size_t)gid * KIN);
  #pragma unroll
  for (int q = 0; q < 12; ++q) {
    uint4 sv;
    sv.x = pack2(t[8*q+0], t[8*q+1]);
    sv.y = pack2(t[8*q+2], t[8*q+3]);
    sv.z = pack2(t[8*q+4], t[8*q+5]);
    sv.w = pack2(t[8*q+6], t[8*q+7]);
    u4[q] = sv;
  }
}

// ---------------------------------------------------------------------------
// conv1: 3x3, 83->64, ReLU, implicit GEMM via WMMA.  One wave produces a
// 32(x) x 64(oc) tile; each B fragment (2 b128 loads) feeds 2 WMMAs.
// ---------------------------------------------------------------------------
__global__ void __launch_bounds__(256)
conv1_kernel(const uint16_t* __restrict__ uin, const uint16_t* __restrict__ wf,
             const float* __restrict__ ub1, uint16_t* __restrict__ h) {
  int wid  = (blockIdx.x * 256 + (int)threadIdx.x) >> 5;
  int lane = threadIdx.x & 31;
  int half = lane >> 4, m = lane & 15;
  int b  = wid >> 7;              // 128 tiles per batch image
  int r2 = wid & 127;
  int y  = r2 >> 1;
  int x0 = (r2 & 1) << 5;         // 32 x-positions per wave

  const uint16_t* ub = uin + (size_t)b * POS * KIN;
  v8f acc[8] = {};                // [mhalf*4 + nt]

  for (int ky = 0; ky < 3; ++ky) {
    int yy = y + ky - 1;
    if (yy < 0 || yy >= H1) continue;            // wave-uniform branch
    for (int kx = 0; kx < 3; ++kx) {
      int xxl = x0 + m + kx - 1;                 // GEMM rows 0..15
      int xxh = x0 + 16 + m + kx - 1;            // GEMM rows 16..31
      bool vl = (xxl >= 0 && xxl < W1);
      bool vh = (xxh >= 0 && xxh < W1);
      const uint16_t* rl = ub + (size_t)(yy * W1 + (vl ? xxl : 0)) * KIN;
      const uint16_t* rh = ub + (size_t)(yy * W1 + (vh ? xxh : 0)) * KIN;
      const uint16_t* wrow = wf + ((size_t)(ky * 3 + kx) * 64 + m) * KIN;
      #pragma unroll
      for (int kc = 0; kc < 3; ++kc) {
        ABf aL, aH;
        zeroF(aL);
        zeroF(aH);
        if (vl) loadA(aL, rl + kc * 32, half);
        if (vh) loadA(aH, rh + kc * 32, half);
        #pragma unroll
        for (int nt = 0; nt < 4; ++nt) {
          ABf bb;
          loadB(bb, wrow + (size_t)(nt * 16) * KIN + kc * 32, half);
          acc[nt] = __builtin_amdgcn_wmma_f32_16x16x32_bf16(
              false, aL.h, false, bb.h, (short)0, acc[nt], false, false);
          acc[4 + nt] = __builtin_amdgcn_wmma_f32_16x16x32_bf16(
              false, aH.h, false, bb.h, (short)0, acc[4 + nt], false, false);
        }
      }
    }
  }

  #pragma unroll
  for (int hr = 0; hr < 2; ++hr) {
    #pragma unroll
    for (int nt = 0; nt < 4; ++nt) {
      float bias = ub1[nt * 16 + m];             // N (=oc) lives on lane&15
      #pragma unroll
      for (int r = 0; r < 8; ++r) {
        float v = fmaxf(acc[hr * 4 + nt][r] + bias, 0.f);
        h[(size_t)(b * POS + y * W1 + x0 + hr * 16 + half * 8 + r) * HID
          + nt * 16 + m] = f2bf(v);
      }
    }
  }
}

// ---------------------------------------------------------------------------
// conv2: 3x3, 64->2, flow += dflow.  uw2 staged in LDS; h read as b128.
// ---------------------------------------------------------------------------
__global__ void __launch_bounds__(256)
conv2_kernel(const uint16_t* __restrict__ h, const float* __restrict__ uw2,
             const float* __restrict__ ub2, float* __restrict__ flow) {
  __shared__ float wl[2 * 9 * 64];               // [c][tap][ic], 4.6 KB
  for (int i = threadIdx.x; i < 2 * 9 * 64; i += blockDim.x) {
    int c = i / 576, r = i - c * 576;
    int tap = r >> 6, ic = r & 63;
    wl[i] = uw2[(c * 64 + ic) * 9 + tap];
  }
  __syncthreads();

  int gid = blockIdx.x * blockDim.x + threadIdx.x;
  int b = gid >> 12, pos = gid & 4095;
  int y = pos >> 6, xq = pos & 63;
  float a0 = ub2[0], a1 = ub2[1];
  for (int ky = 0; ky < 3; ++ky) {
    int yy = y + ky - 1;
    if (yy < 0 || yy >= H1) continue;
    for (int kx = 0; kx < 3; ++kx) {
      int xx = xq + kx - 1;
      if (xx < 0 || xx >= W1) continue;
      int tap = ky * 3 + kx;
      const uint4* hq = (const uint4*)(h + (size_t)(b * POS + yy * W1 + xx) * HID);
      const float* w0 = wl + tap * 64;
      const float* w1 = wl + 576 + tap * 64;
      #pragma unroll
      for (int q = 0; q < 8; ++q) {
        uint4 d = hq[q];
        int ic = q * 8;
        float h0 = bf2f((uint16_t)d.x), h1 = bf2f((uint16_t)(d.x >> 16));
        float h2 = bf2f((uint16_t)d.y), h3 = bf2f((uint16_t)(d.y >> 16));
        float h4 = bf2f((uint16_t)d.z), h5 = bf2f((uint16_t)(d.z >> 16));
        float h6 = bf2f((uint16_t)d.w), h7 = bf2f((uint16_t)(d.w >> 16));
        a0 = fmaf(h0, w0[ic+0], a0); a1 = fmaf(h0, w1[ic+0], a1);
        a0 = fmaf(h1, w0[ic+1], a0); a1 = fmaf(h1, w1[ic+1], a1);
        a0 = fmaf(h2, w0[ic+2], a0); a1 = fmaf(h2, w1[ic+2], a1);
        a0 = fmaf(h3, w0[ic+3], a0); a1 = fmaf(h3, w1[ic+3], a1);
        a0 = fmaf(h4, w0[ic+4], a0); a1 = fmaf(h4, w1[ic+4], a1);
        a0 = fmaf(h5, w0[ic+5], a0); a1 = fmaf(h5, w1[ic+5], a1);
        a0 = fmaf(h6, w0[ic+6], a0); a1 = fmaf(h6, w1[ic+6], a1);
        a0 = fmaf(h7, w0[ic+7], a0); a1 = fmaf(h7, w1[ic+7], a1);
      }
    }
  }
  flow[((size_t)b * 2 + 0) * POS + pos] += a0;
  flow[((size_t)b * 2 + 1) * POS + pos] += a1;
}

// ---------------------------------------------------------------------------
// Upsample flow (half-pixel bilinear, x4), warp img (zeros padding),
// concat sd1 -> feat [b][5][256][256].
// ---------------------------------------------------------------------------
__global__ void __launch_bounds__(256)
warp_kernel(const float* __restrict__ x, const float* __restrict__ flow,
            float* __restrict__ feat) {
  int gid = blockIdx.x * blockDim.x + threadIdx.x;   // BB*HH*WW
  int b = gid >> 16, pos = gid & 65535;
  int y = pos >> 8, xq = pos & 255;

  float sx = xq * 0.25f - 0.375f, sy = y * 0.25f - 0.375f;
  float xf = floorf(sx), yf = floorf(sy);
  float wx = sx - xf, wy = sy - yf;
  int X0 = (int)xf, Y0 = (int)yf;
  int x0c = X0 < 0 ? 0 : (X0 > W1 - 1 ? W1 - 1 : X0);
  int x1c = X0 + 1 < 0 ? 0 : (X0 + 1 > W1 - 1 ? W1 - 1 : X0 + 1);
  int y0c = Y0 < 0 ? 0 : (Y0 > H1 - 1 ? H1 - 1 : Y0);
  int y1c = Y0 + 1 < 0 ? 0 : (Y0 + 1 > H1 - 1 ? H1 - 1 : Y0 + 1);
  const float* F = flow + (size_t)b * 2 * POS;
  float fup[2];
  #pragma unroll
  for (int c = 0; c < 2; ++c) {
    const float* Fc = F + c * POS;
    float v = (Fc[y0c * W1 + x0c] * (1.f - wx) + Fc[y0c * W1 + x1c] * wx) * (1.f - wy)
            + (Fc[y1c * W1 + x0c] * (1.f - wx) + Fc[y1c * W1 + x1c] * wx) * wy;
    fup[c] = v * 4.f;
  }

  float xs = xq + fup[0], ys = y + fup[1];
  float xf2 = floorf(xs), yf2 = floorf(ys);
  float wx1 = xs - xf2, wy1 = ys - yf2;
  int xi = (int)xf2, yi = (int)yf2;
  #pragma unroll
  for (int c = 0; c < 3; ++c) {
    const float* im = x + ((size_t)b * 9 + c) * HH * WW;
    float v00 = 0.f, v10 = 0.f, v01 = 0.f, v11 = 0.f;
    if ((unsigned)yi < (unsigned)HH) {
      if ((unsigned)xi       < (unsigned)WW) v00 = im[yi * WW + xi];
      if ((unsigned)(xi + 1) < (unsigned)WW) v10 = im[yi * WW + xi + 1];
    }
    if ((unsigned)(yi + 1) < (unsigned)HH) {
      if ((unsigned)xi       < (unsigned)WW) v01 = im[(yi + 1) * WW + xi];
      if ((unsigned)(xi + 1) < (unsigned)WW) v11 = im[(yi + 1) * WW + xi + 1];
    }
    feat[((size_t)b * 5 + c) * HH * WW + pos] =
        (v00 * (1.f - wx1) + v10 * wx1) * (1.f - wy1) +
        (v01 * (1.f - wx1) + v11 * wx1) * wy1;
  }
  feat[((size_t)b * 5 + 3) * HH * WW + pos] = x[((size_t)b * 9 + 5) * HH * WW + pos];
  feat[((size_t)b * 5 + 4) * HH * WW + pos] = x[((size_t)b * 9 + 6) * HH * WW + pos];
}

// ---------------------------------------------------------------------------
// Final 3x3 conv 5->64 + BatchNorm (inference) + SiLU -> d_out.
// ---------------------------------------------------------------------------
__global__ void __launch_bounds__(256)
final_kernel(const float* __restrict__ feat, const float* __restrict__ cw,
             const float* __restrict__ cb, const float* __restrict__ gamma,
             const float* __restrict__ beta, const float* __restrict__ rm,
             const float* __restrict__ rv, float* __restrict__ out) {
  int gid = blockIdx.x * blockDim.x + threadIdx.x;   // BB*HH*WW
  int b = gid >> 16, pos = gid & 65535;
  int y = pos >> 8, xq = pos & 255;

  float v[45];
  #pragma unroll
  for (int ic = 0; ic < 5; ++ic) {
    const float* fp = feat + ((size_t)b * 5 + ic) * HH * WW;
    #pragma unroll
    for (int ky = 0; ky < 3; ++ky) {
      int yy = y + ky - 1;
      #pragma unroll
      for (int kx = 0; kx < 3; ++kx) {
        int xx = xq + kx - 1;
        v[ic * 9 + ky * 3 + kx] =
            ((unsigned)yy < (unsigned)HH && (unsigned)xx < (unsigned)WW)
                ? fp[yy * WW + xx] : 0.f;
      }
    }
  }
  for (int oc = 0; oc < 64; ++oc) {
    float acc = cb[oc];
    const float* w = cw + oc * 45;
    #pragma unroll
    for (int k = 0; k < 45; ++k) acc = fmaf(v[k], w[k], acc);
    float z = (acc - rm[oc]) * rsqrtf(rv[oc] + 1e-5f) * gamma[oc] + beta[oc];
    out[((size_t)b * 64 + oc) * HH * WW + pos] = z / (1.f + __expf(-z));
  }
}

// ---------------------------------------------------------------------------
extern "C" void kernel_launch(void* const* d_in, const int* in_sizes, int n_in,
                              void* d_out, int out_size, void* d_ws, size_t ws_size,
                              hipStream_t stream) {
  (void)in_sizes; (void)n_in; (void)out_size; (void)ws_size;
  const float* x      = (const float*)d_in[0];
  const float* enc_w  = (const float*)d_in[1];
  const float* enc_b  = (const float*)d_in[2];
  const float* uw1    = (const float*)d_in[3];
  const float* ub1    = (const float*)d_in[4];
  const float* uw2    = (const float*)d_in[5];
  const float* ub2    = (const float*)d_in[6];
  const float* conv_w = (const float*)d_in[7];
  const float* conv_b = (const float*)d_in[8];
  const float* gamma  = (const float*)d_in[9];
  const float* beta   = (const float*)d_in[10];
  const float* rmean  = (const float*)d_in[11];
  const float* rvar   = (const float*)d_in[12];

  // Workspace carve-out (~150 MB; bf16 corr volume keeps the 20-iteration
  // gather phase resident in the 192 MB L2).
  char* ws = (char*)d_ws;
  size_t off = 0;
  auto carve = [&](size_t bytes) -> char* {
    char* p = ws + off;
    off += (bytes + 255) & ~(size_t)255;
    return p;
  };
  uint16_t* fA   = (uint16_t*)carve((size_t)BB * POS * FD * 2);
  uint16_t* fB   = (uint16_t*)carve((size_t)BB * POS * FD * 2);
  uint16_t* wf   = (uint16_t*)carve((size_t)9 * 64 * KIN * 2);
  uint16_t* cor  = (uint16_t*)carve((size_t)BB * POS * POS * 2);
  uint16_t* uin  = (uint16_t*)carve((size_t)BB * POS * KIN * 2);
  uint16_t* hbuf = (uint16_t*)carve((size_t)BB * POS * HID * 2);
  float*    flow = (float*)  carve((size_t)BB * 2 * POS * 4);
  float*    feat = (float*)  carve((size_t)BB * 5 * HH * WW * 4);

  (void)hipMemsetAsync(flow, 0, (size_t)BB * 2 * POS * 4, stream);

  dim3 blk(256);
  enc_kernel<<<dim3(BB * POS / 256, 2), blk, 0, stream>>>(x, enc_w, enc_b, fA, fB);
  wf_kernel<<<dim3((9 * 64 * KIN + 255) / 256), blk, 0, stream>>>(uw1, wf);
  // BB * 8192 waves (32x64 tiles), 8 waves/block
  corr_kernel<<<dim3(BB * 8192 / 8), blk, 0, stream>>>(fA, fB, cor);

  for (int it = 0; it < ITERS; ++it) {
    sample_kernel<<<dim3(BB * POS / 256), blk, 0, stream>>>(cor, flow, uin);
    // BB * 128 waves (32x64 tiles), 8 waves/block
    conv1_kernel<<<dim3(BB * 128 / 8), blk, 0, stream>>>(uin, wf, ub1, hbuf);
    conv2_kernel<<<dim3(BB * POS / 256), blk, 0, stream>>>(hbuf, uw2, ub2, flow);
  }

  warp_kernel<<<dim3(BB * HH * WW / 256), blk, 0, stream>>>(x, flow, feat);
  final_kernel<<<dim3(BB * HH * WW / 256), blk, 0, stream>>>(
      feat, conv_w, conv_b, gamma, beta, rmean, rvar, (float*)d_out);
}
